// Img_Trans_Multi_Graph_Net_893353197957
// MI455X (gfx1250) — matrix operand order repaired and checked by reading, not compiled, not run on hardware
//
#include <hip/hip_runtime.h>
#include <hip/hip_bf16.h>
#include <math.h>

// ---------------------------------------------------------------------------
// CDNA5 (gfx1250) implementation. bf16 WMMA (v_wmma_f32_16x16x32_bf16) for all
// GEMMs + flash attention; f32 accumulate. wave32; 32x32 register-blocked
// GEMM tile per wave (2x2 WMMA tiles, 1:1 load:wmma ratio).
// ---------------------------------------------------------------------------

typedef __attribute__((ext_vector_type(16))) __bf16 v16bf;
typedef __attribute__((ext_vector_type(8)))  float  v8f;

union FragAB { v16bf v; unsigned int w[8]; };
union FragC  { v8f  v; float f[8]; };

// Single v_perm_b32: pack {hi16(y), hi16(x)} (bf16 truncation).
__device__ __forceinline__ unsigned int pack2_bf16(float x, float y) {
  return __builtin_amdgcn_perm(__float_as_uint(y), __float_as_uint(x), 0x07060302u);
}

__device__ __forceinline__ float clip1000(float x) {
  return fminf(fmaxf(x, -1000.0f), 1000.0f);
}

// A-fragment K offset for VGPR v, lane-half hf (ISA 7.12.2, 16-bit A 16x32)
__device__ __forceinline__ int ka_off(int v, int hf) {
  return (v < 4) ? (hf * 8 + 2 * v) : (16 + hf * 8 + 2 * (v - 4));
}
// B-fragment K offset (same K order as A fragment)
__device__ __forceinline__ int kb_off(int v, int hf) { return (v < 4) ? (hf * 8 + 2 * v) : (16 + hf * 8 + 2 * (v - 4)); }

// ---------------------------------------------------------------------------
// GEMM: C[M,Nd] = act(A[M,K] @ W[Nd,K]^T + bias), optional clip.
// One wave per 32x32 output tile (2x2 WMMA frags). M,Nd multiples of 32,
// K multiple of 32. act: 0=none 1=relu 2=lrelu(0.2)
// ---------------------------------------------------------------------------
__global__ __launch_bounds__(256)
void wmma_gemm(const float* __restrict__ A, const float* __restrict__ W,
               const float* __restrict__ bias, float* __restrict__ C,
               int M, int Nd, int K, int act, int clip) {
  int wave = blockIdx.x * (blockDim.x >> 5) + (threadIdx.x >> 5);
  int n2 = Nd >> 5;
  if (wave >= (M >> 5) * n2) return;
  int lane = threadIdx.x & 31;
  int hf = lane >> 4, mr = lane & 15;
  int tm = (wave / n2) * 32, tn = (wave % n2) * 32;

  const float* ar[2] = { A + (size_t)(tm + mr) * K,      A + (size_t)(tm + 16 + mr) * K };
  const float* wr[2] = { W + (size_t)(tn + mr) * K,      W + (size_t)(tn + 16 + mr) * K };

  FragC cc[2][2];
#pragma unroll
  for (int i = 0; i < 2; i++)
#pragma unroll
    for (int j = 0; j < 2; j++)
#pragma unroll
      for (int r = 0; r < 8; r++) cc[i][j].f[r] = 0.0f;

  for (int kb = 0; kb < K; kb += 32) {
    // prefetch next K block into cache while WMMAs run (global_prefetch_b8)
    if (kb + 32 < K) {
      __builtin_prefetch(ar[0] + kb + 32, 0, 3);
      __builtin_prefetch(ar[1] + kb + 32, 0, 3);
      __builtin_prefetch(wr[0] + kb + 32, 0, 3);
      __builtin_prefetch(wr[1] + kb + 32, 0, 3);
    }
    FragAB a[2], b[2];
#pragma unroll
    for (int v = 0; v < 8; v++) {
      int ka = kb + ka_off(v, hf);
      int kw = kb + kb_off(v, hf);
      a[0].w[v] = pack2_bf16(ar[0][ka], ar[0][ka + 1]);
      a[1].w[v] = pack2_bf16(ar[1][ka], ar[1][ka + 1]);
      b[0].w[v] = pack2_bf16(wr[0][kw], wr[0][kw + 1]);
      b[1].w[v] = pack2_bf16(wr[1][kw], wr[1][kw + 1]);
    }
#pragma unroll
    for (int i = 0; i < 2; i++)
#pragma unroll
      for (int j = 0; j < 2; j++)
        cc[i][j].v = __builtin_amdgcn_wmma_f32_16x16x32_bf16(
            false, a[i].v, false, b[j].v, (short)0, cc[i][j].v, false, false);
  }

#pragma unroll
  for (int j = 0; j < 2; j++) {
    int col = tn + j * 16 + mr;
    float bv = bias ? bias[col] : 0.0f;
#pragma unroll
    for (int i = 0; i < 2; i++) {
#pragma unroll
      for (int r = 0; r < 8; r++) {
        int row = tm + i * 16 + hf * 8 + r;
        float x = cc[i][j].f[r] + bv;
        if (act == 1) x = fmaxf(x, 0.0f);
        else if (act == 2) x = (x > 0.0f) ? x : 0.2f * x;
        if (clip) x = clip1000(x);
        C[(size_t)row * Nd + col] = x;
      }
    }
  }
}

// ---------------------------------------------------------------------------
// Causal flash attention, S rows, H=4 heads, DH=64, D=256.
// qkv: [S, 768] (q|k|v). O: [S,256]. One wave per (head, 16-query block),
// streaming key blocks of 32 with online softmax; P transposed via LDS.
// ---------------------------------------------------------------------------
__global__ __launch_bounds__(256)
void flash_attn(const float* __restrict__ qkv, float* __restrict__ O, int S) {
  int wave = blockIdx.x * 8 + (threadIdx.x >> 5);
  int nrb = S >> 4;
  if (wave >= nrb * 4) return;
  int lane = threadIdx.x & 31;
  int hf = lane >> 4, mr = lane & 15;
  int h = wave / nrb;
  int qbase = (wave % nrb) * 16;
  const int LD3 = 768;
  const float scale = 0.125f;  // 1/sqrt(64)

  extern __shared__ float lds[];
  float* Pl = lds + (threadIdx.x >> 5) * (16 * 32);

  // Q A-fragments for d-chunks 0-31 and 32-63 (pre-scaled)
  FragAB qa[2];
  const float* qrow = qkv + (size_t)(qbase + mr) * LD3 + h * 64;
#pragma unroll
  for (int ch = 0; ch < 2; ch++)
#pragma unroll
    for (int v = 0; v < 8; v++) {
      int d = ch * 32 + ka_off(v, hf);
      qa[ch].w[v] = pack2_bf16(qrow[d] * scale, qrow[d + 1] * scale);
    }

  float m[8], l[8];
  FragC o[4];
#pragma unroll
  for (int r = 0; r < 8; r++) { m[r] = -3.0e38f; l[r] = 0.0f; }
#pragma unroll
  for (int t = 0; t < 4; t++)
#pragma unroll
    for (int r = 0; r < 8; r++) o[t].f[r] = 0.0f;

  int kend = qbase + 16;
  for (int ks = 0; ks < kend; ks += 32) {
    // ---- scores: 16 queries x 32 keys, two 16x16 tiles ----
    FragC sc[2];
#pragma unroll
    for (int t = 0; t < 2; t++) {
#pragma unroll
      for (int r = 0; r < 8; r++) sc[t].f[r] = 0.0f;
      const float* krow = qkv + (size_t)(ks + t * 16 + mr) * LD3 + 256 + h * 64;
#pragma unroll
      for (int ch = 0; ch < 2; ch++) {
        FragAB kf;
#pragma unroll
        for (int v = 0; v < 8; v++) {
          int d = ch * 32 + kb_off(v, hf);
          kf.w[v] = pack2_bf16(krow[d], krow[d + 1]);
        }
        sc[t].v = __builtin_amdgcn_wmma_f32_16x16x32_bf16(false, qa[ch].v, false, kf.v,
                                                          (short)0, sc[t].v, false, false);
      }
      // causal mask
      int key = ks + t * 16 + mr;
#pragma unroll
      for (int r = 0; r < 8; r++)
        if (key > qbase + hf * 8 + r) sc[t].f[r] = -3.0e38f;
    }

    // ---- online softmax (rows live in 16-lane groups) ----
#pragma unroll
    for (int r = 0; r < 8; r++) {
      float mx = fmaxf(sc[0].f[r], sc[1].f[r]);
      for (int dlt = 8; dlt >= 1; dlt >>= 1) mx = fmaxf(mx, __shfl_xor(mx, dlt, 32));
      float mnew = fmaxf(m[r], mx);
      float corr = __expf(m[r] - mnew);
      m[r] = mnew;
      float p0 = __expf(sc[0].f[r] - mnew);
      float p1 = __expf(sc[1].f[r] - mnew);
      float rs = p0 + p1;
      for (int dlt = 8; dlt >= 1; dlt >>= 1) rs += __shfl_xor(rs, dlt, 32);
      l[r] = l[r] * corr + rs;
#pragma unroll
      for (int t = 0; t < 4; t++) o[t].f[r] *= corr;
      int row = hf * 8 + r;
      Pl[row * 32 + mr] = p0;        // C-layout -> LDS [row][key]
      Pl[row * 32 + 16 + mr] = p1;
    }
    asm volatile("s_wait_dscnt 0" ::: "memory");  // same-wave LDS ordering

    // ---- O += P(16x32) @ V(32x64) ----
    FragAB pa;
#pragma unroll
    for (int v = 0; v < 8; v++) {
      int k0 = ka_off(v, hf);
      pa.w[v] = pack2_bf16(Pl[mr * 32 + k0], Pl[mr * 32 + k0 + 1]);
    }
#pragma unroll
    for (int t = 0; t < 4; t++) {
      FragAB vb;
#pragma unroll
      for (int v = 0; v < 8; v++) {
        int kk = kb_off(v, hf);  // key within block
        const float* vp = qkv + (size_t)(ks + kk) * LD3 + 512 + h * 64 + t * 16 + mr;
        vb.w[v] = pack2_bf16(vp[0], vp[LD3]);
      }
      o[t].v = __builtin_amdgcn_wmma_f32_16x16x32_bf16(false, pa.v, false, vb.v,
                                                       (short)0, o[t].v, false, false);
    }
  }

#pragma unroll
  for (int t = 0; t < 4; t++) {
    int col = h * 64 + t * 16 + mr;
#pragma unroll
    for (int r = 0; r < 8; r++) {
      int row = qbase + hf * 8 + r;
      O[(size_t)row * 256 + col] = o[t].f[r] / l[r];
    }
  }
}

// ---------------------------------------------------------------------------
// x = LayerNorm(x + res) * g + b, D=256, one block (256 thr) per row
// ---------------------------------------------------------------------------
__global__ __launch_bounds__(256)
void ln_residual(float* __restrict__ x, const float* __restrict__ res,
                 const float* __restrict__ g, const float* __restrict__ b) {
  const int D = 256;
  int row = blockIdx.x, c = threadIdx.x;
  float v = x[(size_t)row * D + c] + res[(size_t)row * D + c];
  __shared__ float red[256];
  red[c] = v; __syncthreads();
  for (int s = 128; s > 0; s >>= 1) { if (c < s) red[c] += red[c + s]; __syncthreads(); }
  float mean = red[0] * (1.0f / D);
  __syncthreads();
  float d = v - mean;
  red[c] = d * d; __syncthreads();
  for (int s = 128; s > 0; s >>= 1) { if (c < s) red[c] += red[c + s]; __syncthreads(); }
  float var = red[0] * (1.0f / D);
  x[(size_t)row * D + c] = d * rsqrtf(var + 1e-5f) * g[c] + b[c];
}

__global__ void clip_kernel(float* x, int n) {
  int i = blockIdx.x * blockDim.x + threadIdx.x;
  if (i < n) x[i] = clip1000(x[i]);
}

__global__ void fill_kernel(float* x, int n, float v) {
  int i = blockIdx.x * blockDim.x + threadIdx.x;
  if (i < n) x[i] = v;
}

__global__ void fix_m(float* m, int n) {  // empty segments: -inf -> 0
  int i = blockIdx.x * blockDim.x + threadIdx.x;
  if (i < n && m[i] < -1.0e37f) m[i] = 0.0f;
}

__global__ void copy_h_to_z(const float* __restrict__ h, float* __restrict__ z, int Nn) {
  int i = blockIdx.x * blockDim.x + threadIdx.x;
  if (i >= Nn * 256) return;
  int r = i >> 8, c = i & 255;
  z[(size_t)r * 512 + 256 + c] = h[i];
}

// ---------------------------------------------------------------------------
// TransformerConv edge kernels (GH=4 heads x GC=32 channels)
// ---------------------------------------------------------------------------
__device__ __forceinline__ void atomicMaxFloat(float* addr, float val) {
  if (val >= 0.0f) atomicMax((int*)addr, __float_as_int(val));
  else             atomicMin((unsigned int*)addr, __float_as_uint(val));
}

__global__ void edge_alpha_max(const int* __restrict__ src, const int* __restrict__ dst,
                               const float* __restrict__ q, const float* __restrict__ k,
                               float* __restrict__ alpha, float* __restrict__ m, int E) {
  int i = blockIdx.x * blockDim.x + threadIdx.x;
  if (i >= E * 4) return;
  int e = i >> 2, hh = i & 3;
  const float* qp = q + (size_t)dst[e] * 128 + hh * 32;
  const float* kp = k + (size_t)src[e] * 128 + hh * 32;
  float acc = 0.0f;
#pragma unroll
  for (int c = 0; c < 32; c++) acc += qp[c] * kp[c];
  acc *= 0.17677669529663687f;  // 1/sqrt(32)
  alpha[i] = acc;
  atomicMaxFloat(&m[dst[e] * 4 + hh], acc);
}

__global__ void edge_exp_sum(const int* __restrict__ dst, float* __restrict__ alpha,
                             const float* __restrict__ m, float* __restrict__ ssum, int E) {
  int i = blockIdx.x * blockDim.x + threadIdx.x;
  if (i >= E * 4) return;
  int e = i >> 2, hh = i & 3;
  float v = __expf(alpha[i] - m[dst[e] * 4 + hh]);
  alpha[i] = v;
  atomicAdd(&ssum[dst[e] * 4 + hh], v);
}

__global__ void edge_scatter(const int* __restrict__ src, const int* __restrict__ dst,
                             const float* __restrict__ alpha, const float* __restrict__ ssum,
                             const float* __restrict__ v, float* __restrict__ agg, int E) {
  int i = blockIdx.x * blockDim.x + threadIdx.x;
  if (i >= E * 4) return;
  int e = i >> 2, hh = i & 3;
  float a = alpha[i] / (ssum[dst[e] * 4 + hh] + 1e-16f);
  const float* vp = v + (size_t)src[e] * 128 + hh * 32;
  float* op = agg + (size_t)dst[e] * 128 + hh * 32;
#pragma unroll
  for (int c = 0; c < 32; c++) atomicAdd(&op[c], a * vp[c]);
}

// Instance-norm over nodes per channel, then lrelu+clip, write into z column slice
__global__ __launch_bounds__(256)
void instnorm_to_z(const float* __restrict__ agg, float* __restrict__ z, int zoff, int Nn) {
  int ch = blockIdx.x;  // 0..127
  __shared__ float r1[256], r2[256];
  float s1 = 0.0f, s2 = 0.0f;
  for (int r = threadIdx.x; r < Nn; r += 256) {
    float v = agg[(size_t)r * 128 + ch];
    s1 += v; s2 += v * v;
  }
  r1[threadIdx.x] = s1; r2[threadIdx.x] = s2; __syncthreads();
  for (int s = 128; s > 0; s >>= 1) {
    if (threadIdx.x < s) { r1[threadIdx.x] += r1[threadIdx.x + s]; r2[threadIdx.x] += r2[threadIdx.x + s]; }
    __syncthreads();
  }
  float mean = r1[0] / Nn;
  float var = fmaxf(r2[0] / Nn - mean * mean, 0.0f);
  float inv = rsqrtf(var + 1e-5f);
  for (int r = threadIdx.x; r < Nn; r += 256) {
    float v = (agg[(size_t)r * 128 + ch] - mean) * inv;
    v = (v > 0.0f) ? v : 0.2f * v;
    z[(size_t)r * 512 + zoff + ch] = clip1000(v);
  }
}

// fc2 (128->2) + softmax; out = [logits (N*2) | probs (N*2)]
__global__ void fc2_softmax(const float* __restrict__ x, const float* __restrict__ W,
                            const float* __restrict__ b, float* __restrict__ out, int Nn) {
  int r = blockIdx.x * blockDim.x + threadIdx.x;
  if (r >= Nn) return;
  float a0 = b[0], a1 = b[1];
  const float* xr = x + (size_t)r * 128;
#pragma unroll 4
  for (int c = 0; c < 128; c++) { a0 += xr[c] * W[c]; a1 += xr[c] * W[128 + c]; }
  out[r * 2] = a0; out[r * 2 + 1] = a1;
  float mx = fmaxf(a0, a1);
  float e0 = __expf(a0 - mx), e1 = __expf(a1 - mx);
  float s = e0 + e1;
  out[(size_t)Nn * 2 + r * 2] = e0 / s;
  out[(size_t)Nn * 2 + r * 2 + 1] = e1 / s;
}

// ---------------------------------------------------------------------------
// Host orchestration
// Input order (recursive dict order): 0 x, 1 edge_index, 2 img_feat,
// 3 video_adj_list, 4.. params leaves (see mapping in comments below).
// ---------------------------------------------------------------------------
extern "C" void kernel_launch(void* const* d_in, const int* in_sizes, int n_in,
                              void* d_out, int out_size, void* d_ws, size_t ws_size,
                              hipStream_t stream) {
  (void)out_size; (void)ws_size;
  const float* img_feat = (const float*)d_in[2];
  const int* vadj = (const int*)d_in[3];
  const int E = in_sizes[3] / 2;
  const int N = 4096;

  const float* Pp[64];
  for (int i = 0; i < n_in && i < 64; i++) Pp[i] = (const float*)d_in[i];

  // workspace layout (floats); total 14,680,064 f32 = 56 MiB (fits L2)
  float* A  = (float*)d_ws;                 // h   [N,256]
  float* B  = A  + (size_t)N * 256;         // qkv [N,768], reused as z [N,512]
  float* C  = B  + (size_t)N * 768;         // t   [N,256]
  float* Db = C  + (size_t)N * 256;         // big [N,2048] (ff / conv scratch)
  float* Ha = Db + (size_t)N * 2048;        // ha  [N,256]

  auto gemm = [&](const float* Ain, const float* W, const float* bias, float* Cout,
                  int M, int Nd, int K, int act, int clip) {
    int waves = (M >> 5) * (Nd >> 5);
    wmma_gemm<<<(waves + 7) / 8, 256, 0, stream>>>(Ain, W, bias, Cout, M, Nd, K, act, clip);
  };

  auto conv = [&](const float* X, int pb, int zoff) {
    float* Q   = Db;
    float* Kk  = Db + (size_t)N * 128;
    float* V   = Db + (size_t)N * 256;
    float* Agg = Db + (size_t)N * 384;
    float* Al  = Db + (size_t)N * 512;       // [E,4]
    float* Mm  = Al + (size_t)E * 4;         // [N,4]
    float* Ss  = Mm + (size_t)N * 4;         // [N,4]
    gemm(X, Pp[pb + 0], Pp[pb + 1], Q,   N, 128, 256, 0, 0);
    gemm(X, Pp[pb + 2], Pp[pb + 3], Kk,  N, 128, 256, 0, 0);
    gemm(X, Pp[pb + 4], Pp[pb + 5], V,   N, 128, 256, 0, 0);
    gemm(X, Pp[pb + 6], Pp[pb + 7], Agg, N, 128, 256, 0, 0);  // skip preload: agg = s(x)
    fill_kernel<<<(N * 4 + 255) / 256, 256, 0, stream>>>(Mm, N * 4, -3.0e38f);
    fill_kernel<<<(N * 4 + 255) / 256, 256, 0, stream>>>(Ss, N * 4, 0.0f);
    const int* src = vadj; const int* dst = vadj + E;
    int tE = E * 4;
    edge_alpha_max<<<(tE + 255) / 256, 256, 0, stream>>>(src, dst, Q, Kk, Al, Mm, E);
    fix_m<<<(N * 4 + 255) / 256, 256, 0, stream>>>(Mm, N * 4);
    edge_exp_sum<<<(tE + 255) / 256, 256, 0, stream>>>(dst, Al, Mm, Ss, E);
    edge_scatter<<<(tE + 255) / 256, 256, 0, stream>>>(src, dst, Al, Ss, V, Agg, E);
    instnorm_to_z<<<128, 256, 0, stream>>>(Agg, B, zoff, N);
  };

  // 1) h = img_fc(img_feat)                     params: 4=W[256,2048], 5=b
  gemm(img_feat, Pp[4], Pp[5], A, N, 256, 2048, 0, 0);

  // 2) two encoder layers. per-layer leaf base = 6 + 12*L:
  //    +0 attn_in.W +1 b, +2 attn_out.W +3 b, +4 ff1.W +5 b, +6 ff2.W +7 b,
  //    +8 g1 +9 b1, +10 g2 +11 b2
  for (int L = 0; L < 2; L++) {
    int b0 = 6 + 12 * L;
    gemm(A, Pp[b0 + 0], Pp[b0 + 1], B, N, 768, 256, 0, 0);            // qkv
    flash_attn<<<(N / 16 * 4) / 8, 256, 8 * 16 * 32 * sizeof(float), stream>>>(B, C, N);
    gemm(C, Pp[b0 + 2], Pp[b0 + 3], Db, N, 256, 256, 0, 0);           // proj
    ln_residual<<<N, 256, 0, stream>>>(A, Db, Pp[b0 + 8], Pp[b0 + 9]);
    gemm(A, Pp[b0 + 4], Pp[b0 + 5], Db, N, 2048, 256, 1, 0);          // ff1 + relu
    gemm(Db, Pp[b0 + 6], Pp[b0 + 7], C, N, 256, 2048, 0, 0);          // ff2
    ln_residual<<<N, 256, 0, stream>>>(A, C, Pp[b0 + 10], Pp[b0 + 11]);
  }

  // 3) clip, fusion (clip), h <- fusion(h)      params: 30=fusion.W, 31=b
  clip_kernel<<<(N * 256 + 255) / 256, 256, 0, stream>>>(A, N * 256);
  gemm(A, Pp[30], Pp[31], C, N, 256, 256, 0, 1);
  hipMemcpyAsync(A, C, (size_t)N * 256 * sizeof(float), hipMemcpyDeviceToDevice, stream);

  // 4) ha = clip(MHA(h))                        params: 32/33 attn_in, 34/35 attn_out
  gemm(A, Pp[32], Pp[33], B, N, 768, 256, 0, 0);
  flash_attn<<<(N / 16 * 4) / 8, 256, 8 * 16 * 32 * sizeof(float), stream>>>(B, C, N);
  gemm(C, Pp[34], Pp[35], Ha, N, 256, 256, 0, 1);

  // 5) graph convs -> z cols [0,128) and [128,256); gc_orig=36..43, gc_attn=44..51
  conv(A, 36, 0);
  conv(Ha, 44, 128);

  // 6) z cols [256,512) = h
  copy_h_to_z<<<(N * 256 + 255) / 256, 256, 0, stream>>>(A, B, N);

  // 7) fc1 (lrelu) then fc2 + softmax          params: 52/53 fc1, 54/55 fc2
  gemm(B, Pp[52], Pp[53], C, N, 128, 512, 2, 0);
  fc2_softmax<<<(N + 255) / 256, 256, 0, stream>>>(C, Pp[54], Pp[55], (float*)d_out, N);
}
